// LSTM_68195490726233
// MI455X (gfx1250) — matrix operand order repaired and checked by reading, not compiled, not run on hardware
//
#include <hip/hip_runtime.h>
#include <hip/hip_bf16.h>

// ---- problem constants ----
#define H_DIM 1024
#define S_LEN 128
#define I_DIM 1024
#define NG    4096   // 4*H (gate-interleaved columns)
#define K2    2048   // I + H
#define NKT   (K2 / 32)
#define LDP   40     // LDS row pitch in halves (32 + 8 pad), 80B -> 16B aligned
#define GP    132    // f32 epilogue tile pitch

typedef __attribute__((ext_vector_type(16))) _Float16 v16h;
typedef __attribute__((ext_vector_type(8)))  float    v8f;
typedef __attribute__((ext_vector_type(4)))  float    v4f;
typedef __attribute__((ext_vector_type(4)))  unsigned v4u;   // clean SSA vector for B staging

union H8    { _Float16 h[8];  v4u q; };
union AFrag { v16h v; unsigned u[8];  };
union BFrag { v16h v; v4u q[2];       };

struct SmemAB { _Float16 A[2][128 * LDP]; _Float16 B[2][128 * LDP]; }; // 40 KB
union  Smem   { SmemAB ab; float g[128 * GP]; };                       // 67.6 KB

__device__ __forceinline__ float fast_tanh(float x) {
#if __has_builtin(__builtin_amdgcn_tanhf)
  return __builtin_amdgcn_tanhf(x);        // native v_tanh_f32 on gfx1250
#else
  return tanhf(x);
#endif
}
__device__ __forceinline__ float fast_sigmoid(float x) {
  return 0.5f * fast_tanh(0.5f * x) + 0.5f;
}

// Column mapping (gate-interleaved so one WG tile holds all 4 gates of 32 units):
//   n' = hh_block*128 + g*32 + hh_low,  hh = hh_block*32 + hh_low,  g in {i,f,g,o}
// Weights stored TRANSPOSED: Wt[n'][k], k<1024 -> input weights, k>=1024 -> hidden.
__global__ void lstm_convert_w(const float* __restrict__ wii, const float* __restrict__ wif,
                               const float* __restrict__ wig, const float* __restrict__ wio,
                               const float* __restrict__ whi, const float* __restrict__ whf,
                               const float* __restrict__ whg, const float* __restrict__ who,
                               _Float16* __restrict__ Wt) {
  int idx = blockIdx.x * blockDim.x + threadIdx.x;   // 0 .. NG*K2
  int np  = idx >> 11;            // n' : 0..4095
  int k   = idx & (K2 - 1);       // 0..2047
  int g   = (np >> 5) & 3;
  int hh  = (np >> 7) * 32 + (np & 31);
  const float* b0 = (k < I_DIM) ? wii : whi;
  const float* b1 = (k < I_DIM) ? wif : whf;
  const float* b2 = (k < I_DIM) ? wig : whg;
  const float* b3 = (k < I_DIM) ? wio : who;
  const float* base = (g == 0) ? b0 : (g == 1) ? b1 : (g == 2) ? b2 : b3;
  int kk = (k < I_DIM) ? k : (k - I_DIM);
  Wt[idx] = (_Float16)base[(size_t)kk * H_DIM + hh];
}

__global__ void lstm_bias(const float* __restrict__ bii, const float* __restrict__ bhi,
                          const float* __restrict__ bif, const float* __restrict__ bhf,
                          const float* __restrict__ big, const float* __restrict__ bhg,
                          const float* __restrict__ bio, const float* __restrict__ bho,
                          float* __restrict__ bias) {
  int np = blockIdx.x * blockDim.x + threadIdx.x;    // n' : 0..4095
  int g  = (np >> 5) & 3;
  int hh = (np >> 7) * 32 + (np & 31);
  const float* bi = (g == 0) ? bii : (g == 1) ? bif : (g == 2) ? big : bio;
  const float* bh = (g == 0) ? bhi : (g == 1) ? bhf : (g == 2) ? bhg : bho;
  bias[np] = bi[hh] + bh[hh];
}

__global__ void lstm_init(float* __restrict__ h, float* __restrict__ c) {
  int idx = blockIdx.x * blockDim.x + threadIdx.x;
  h[idx] = 0.0f;
  c[idx] = 0.0f;
}

// One full LSTM timestep: G = [x_t | h_in] @ Wt^T (WMMA f16->f32), bias folded
// into the accumulator init, cell fused in the epilogue via LDS transpose.
// Double-buffered LDS; branch-free pipelined K loop (x-half and h-half split,
// last tile peeled) so every buffer index / source pointer is compile-time.
__launch_bounds__(256)
__global__ void lstm_step(const float* __restrict__ x, const float* __restrict__ h_in,
                          const _Float16* __restrict__ Wt, const float* __restrict__ bias,
                          float* __restrict__ c, float* __restrict__ h_out,
                          float* __restrict__ out_h, float* __restrict__ out_c,
                          int t, int last) {
  __shared__ __align__(16) Smem smem;

  const int tid   = threadIdx.x;
  const int bN    = blockIdx.x;       // 32 tiles over n'
  const int bM    = blockIdx.y;       // 8 tiles over batch rows
  const int lane  = tid & 31;
  const int wv    = tid >> 5;
  const int waveM = wv & 1;           // 2 waves x 64 rows
  const int waveN = wv >> 1;          // 4 waves x 32 cols
  const int lr    = lane & 15;
  const int lh    = lane >> 4;

  // staging slots (2 per thread): A tile 128x32 (f32 src), B tile 128(n')x32(k)
  const int rowA = tid >> 2;          // 0..63 (+64 for slot 1)
  const int ccA  = (tid & 3) * 8;

  // ---- loop-invariant base pointers (hoisted 64-bit address math) ----
  const float*    baseX[2];
  const float*    baseH[2];
  const _Float16* baseW[2];
#pragma unroll
  for (int s = 0; s < 2; ++s) {
    const int row = rowA + s * 64;
    const int gm  = bM * 128 + row;
    baseX[s] = x    + (size_t)gm * (S_LEN * I_DIM) + (size_t)t * I_DIM + ccA;
    baseH[s] = h_in + (size_t)gm * H_DIM + ccA - I_DIM;   // indexed by kb directly
    baseW[s] = Wt   + (size_t)(bN * 128 + row) * K2 + ccA;
  }

  // ---- accumulators initialized with the per-column bias ----
  v8f acc[4][2];
#pragma unroll
  for (int sn = 0; sn < 2; ++sn) {
    const float bcol = bias[bN * 128 + waveN * 32 + sn * 16 + lr];
#pragma unroll
    for (int sm = 0; sm < 4; ++sm) {
      v8f a;
#pragma unroll
      for (int v = 0; v < 8; ++v) a[v] = bcol;
      acc[sm][sn] = a;
    }
  }

  v4f pfA[2][2];
  v4u pfB[2];

  // mode 0: A from x (non-temporal); mode 1: A from h_in; mode constant per call
  auto prefetch = [&](int kt, int mode) {
    const int kb = kt * 32;
#pragma unroll
    for (int s = 0; s < 2; ++s) {
      if (mode == 0) {
        const v4f* src = (const v4f*)(baseX[s] + kb);
        pfA[s][0] = __builtin_nontemporal_load(src);
        pfA[s][1] = __builtin_nontemporal_load(src + 1);
      } else {
        const v4f* src = (const v4f*)(baseH[s] + kb);
        pfA[s][0] = src[0];
        pfA[s][1] = src[1];
      }
      pfB[s] = *(const v4u*)(baseW[s] + kb);
    }
  };

  auto commit = [&](int buf) {       // buf constant at every call site
#pragma unroll
    for (int s = 0; s < 2; ++s) {
      H8 tmp;
      tmp.h[0] = (_Float16)pfA[s][0].x; tmp.h[1] = (_Float16)pfA[s][0].y;
      tmp.h[2] = (_Float16)pfA[s][0].z; tmp.h[3] = (_Float16)pfA[s][0].w;
      tmp.h[4] = (_Float16)pfA[s][1].x; tmp.h[5] = (_Float16)pfA[s][1].y;
      tmp.h[6] = (_Float16)pfA[s][1].z; tmp.h[7] = (_Float16)pfA[s][1].w;
      *(v4u*)&smem.ab.A[buf][(rowA + s * 64) * LDP + ccA] = tmp.q;
      *(v4u*)&smem.ab.B[buf][(rowA + s * 64) * LDP + ccA] = pfB[s];
    }
  };

  // mode 0/1 as above; mode 2: no prefetch/commit (last tile)
  auto kstep = [&](int buf, int ktNext, int mode) {
    if (mode != 2) prefetch(ktNext, mode);

    // A fragment: lanes 0-15 K=0..15, lanes 16-31 K=16..31 (merges into 2x ds_load_b128)
    AFrag a[4];
#pragma unroll
    for (int sm = 0; sm < 4; ++sm) {
      const int r = waveM * 64 + sm * 16 + lr;
#pragma unroll
      for (int v = 0; v < 8; ++v) {
        const int kk = 16 * lh + 2 * v;
        a[sm].u[v] = *(const unsigned*)&smem.ab.A[buf][r * LDP + kk];
      }
    }
    // B fragment: lanes 0-15 K=0..15, lanes 16-31 K=16..31 (2x ds_load_b128)
    BFrag bf[2];
#pragma unroll
    for (int sn = 0; sn < 2; ++sn) {
      const int n = waveN * 32 + sn * 16 + lr;
      const v4u* p = (const v4u*)&smem.ab.B[buf][n * LDP + lh * 16];
      bf[sn].q[0] = p[0];
      bf[sn].q[1] = p[1];
    }

#pragma unroll
    for (int sm = 0; sm < 4; ++sm)
#pragma unroll
      for (int sn = 0; sn < 2; ++sn)
        acc[sm][sn] = __builtin_amdgcn_wmma_f32_16x16x32_f16(
            false, a[sm].v, false, bf[sn].v, (short)0, acc[sm][sn], false, false);

    if (mode != 2) commit(buf ^ 1);  // other buffer: no extra barrier needed
    __syncthreads();
  };

  // ---- branch-free pipelined K loop ----
  prefetch(0, 0);
  commit(0);
  __syncthreads();

#pragma unroll 1
  for (int kt = 0; kt < 30; kt += 2) {   // tiles 0..29, prefetch 1..30 (all x)
    kstep(0, kt + 1, 0);
    kstep(1, kt + 2, 0);
  }
  kstep(0, 31, 0);                        // tile 30, prefetch 31 (x)
  kstep(1, 32, 1);                        // tile 31, prefetch 32 (h)
#pragma unroll 1
  for (int kt = 32; kt < 62; kt += 2) {  // tiles 32..61, prefetch 33..62 (all h)
    kstep(0, kt + 1, 1);
    kstep(1, kt + 2, 1);
  }
  kstep(0, 63, 1);                        // tile 62, prefetch 63 (h)
  kstep(1, 0, 2);                         // tile 63, no prefetch

  // ---- epilogue: transpose accumulators through LDS, apply LSTM cell ----
#pragma unroll
  for (int sm = 0; sm < 4; ++sm) {
    const int m0 = waveM * 64 + sm * 16 + 8 * lh;
#pragma unroll
    for (int sn = 0; sn < 2; ++sn) {
      const int nn = waveN * 32 + sn * 16 + lr;
#pragma unroll
      for (int v = 0; v < 8; ++v) smem.g[(m0 + v) * GP + nn] = acc[sm][sn][v];
    }
  }
  __syncthreads();

#pragma unroll
  for (int e = 0; e < 16; ++e) {
    const int idx2 = e * 256 + tid;       // 0..4095 over (m, hh_low)
    const int m  = idx2 >> 5;             // 0..127
    const int hl = idx2 & 31;
    const float pi = smem.g[m * GP +       hl];
    const float pf = smem.g[m * GP +  32 + hl];
    const float pg = smem.g[m * GP +  64 + hl];
    const float po = smem.g[m * GP +  96 + hl];
    const int    b  = bM * 128 + m;
    const int    hh = bN * 32 + hl;
    const size_t ci = (size_t)b * H_DIM + hh;
    const float ig = fast_sigmoid(pi);
    const float fg = fast_sigmoid(pf);
    const float gg = fast_tanh(pg);
    const float og = fast_sigmoid(po);
    const float cn = fg * c[ci] + ig * gg;
    const float hn = og * fast_tanh(cn);
    c[ci]     = cn;
    h_out[ci] = hn;
    if (last) {
      out_h[ci] = hn;
      out_c[ci] = cn;
    } else {
      // warm L2 for next timestep's input slice (global_prefetch_b8); each
      // element of x[:, t+1, :] is prefetched exactly once across the grid.
      __builtin_prefetch(x + (size_t)b * (S_LEN * I_DIM) + (size_t)(t + 1) * I_DIM + hh, 0, 1);
    }
  }
}

extern "C" void kernel_launch(void* const* d_in, const int* in_sizes, int n_in,
                              void* d_out, int out_size, void* d_ws, size_t ws_size,
                              hipStream_t stream) {
  // setup_inputs() dict order:
  const float* x   = (const float*)d_in[0];
  const float* wii = (const float*)d_in[1];
  const float* whi = (const float*)d_in[2];
  const float* wif = (const float*)d_in[3];
  const float* whf = (const float*)d_in[4];
  const float* wig = (const float*)d_in[5];
  const float* whg = (const float*)d_in[6];
  const float* wio = (const float*)d_in[7];
  const float* who = (const float*)d_in[8];
  const float* bii = (const float*)d_in[9];
  const float* bhi = (const float*)d_in[10];
  const float* bif = (const float*)d_in[11];
  const float* bhf = (const float*)d_in[12];
  const float* big = (const float*)d_in[13];
  const float* bhg = (const float*)d_in[14];
  const float* bio = (const float*)d_in[15];
  const float* bho = (const float*)d_in[16];

  // workspace layout (~28.1 MB total)
  char* ws = (char*)d_ws;
  _Float16* Wt = (_Float16*)ws;                         // 16 MiB  f16 [NG][K2]
  size_t off = (size_t)NG * K2 * sizeof(_Float16);
  float* bias = (float*)(ws + off); off += 64 * 1024;   // 16 KiB used
  float* hA   = (float*)(ws + off); off += (size_t)H_DIM * H_DIM * sizeof(float);
  float* hB   = (float*)(ws + off); off += (size_t)H_DIM * H_DIM * sizeof(float);
  float* cbuf = (float*)(ws + off);

  float* out_h = (float*)d_out;
  float* out_c = out_h + (size_t)H_DIM * H_DIM;

  lstm_convert_w<<<(NG * K2) / 256, 256, 0, stream>>>(wii, wif, wig, wio,
                                                      whi, whf, whg, who, Wt);
  lstm_bias<<<NG / 256, 256, 0, stream>>>(bii, bhi, bif, bhf, big, bhg, bio, bho, bias);
  lstm_init<<<(H_DIM * H_DIM) / 256, 256, 0, stream>>>(hA, cbuf);

  // one fused kernel per timestep; h double-buffered across steps
  dim3 ggrid(NG / 128, H_DIM / 128);   // (32, 8) WGs of 256 threads
  for (int t = 0; t < S_LEN; ++t) {
    const float* hin  = (t & 1) ? hB : hA;
    float*       hout = (t & 1) ? hA : hB;
    lstm_step<<<ggrid, 256, 0, stream>>>(x, hin, Wt, bias, cbuf, hout,
                                         out_h, out_c, t, (t == S_LEN - 1) ? 1 : 0);
  }
}
